// REIN_55628416417817
// MI455X (gfx1250) — compile-verified, matrix operand order checked by reading, not run on hardware
//
#include <hip/hip_runtime.h>
#include <math.h>
#include <stdint.h>

// ---------------------------------------------------------------------------
// MI455X (gfx1250) implementation of the rotation-equivariant ResNet + NetVLAD
// pipeline. Convolutions: BF16 implicit-GEMM on v_wmma_f32_16x16x32_bf16 with
// double-buffered LDS tiles; weight tiles are streamed by the Tensor Data
// Mover (tensor_load_to_lds + s_wait_tensorcnt) overlapped with WMMA compute.
// Everything else (bicubic warps, pooling, norms, VLAD) is bandwidth-trivial.
// ---------------------------------------------------------------------------

typedef __attribute__((ext_vector_type(16))) __bf16 v16bf;
typedef __attribute__((ext_vector_type(8)))  float  v8f;
typedef __attribute__((ext_vector_type(4)))  unsigned int v4u;
typedef __attribute__((ext_vector_type(8)))  int v8i;
typedef __attribute__((ext_vector_type(4)))  int v4i;

struct alignas(16) B128 { unsigned int w[4]; };
union FragU { B128 q[2]; v16bf v; };

#ifndef __has_builtin
#define __has_builtin(x) 0
#endif
#if defined(__AMDGCN__) && __has_builtin(__builtin_amdgcn_tensor_load_to_lds) && \
    __has_builtin(__builtin_amdgcn_s_wait_tensorcnt)
#define USE_TDM 1
#endif

__device__ __forceinline__ unsigned short f2bf(float f) {
  unsigned u = __float_as_uint(f);
  u += 0x7fffu + ((u >> 16) & 1u);      // round-to-nearest-even
  return (unsigned short)(u >> 16);
}
__device__ __forceinline__ float bf2f(unsigned short h) {
  return __uint_as_float(((unsigned)h) << 16);
}

// PyTorch bicubic weights, A = -0.75
__device__ __forceinline__ void cubic_w(float t, float w[4]) {
  const float A = -0.75f;
  float t2 = t * t, t3 = t2 * t;
  w[0] = A * (t3 - 2.f * t2 + t);
  w[1] = (A + 2.f) * t3 - (A + 3.f) * t2 + 1.f;
  float u = 1.f - t, u2 = u * u, u3 = u2 * u;
  w[2] = (A + 2.f) * u3 - (A + 3.f) * u2 + 1.f;
  w[3] = A * (t2 - t3);
}

// ---------------------------------------------------------------------------
// Weight prep: fold BN (scale = g*rsqrt(v+eps)) into BF16 weights laid out as
// WT[n][k], k = (ky*KW+kx)*C + ci, K padded to multiple of 32 with zeros.
// ---------------------------------------------------------------------------
__global__ void prep_wgt(const float* __restrict__ w, const float* __restrict__ g,
                         const float* __restrict__ b, const float* __restrict__ m,
                         const float* __restrict__ v, unsigned short* __restrict__ wt,
                         float* __restrict__ bias, int N, int C, int KH, int KW, int Kpad) {
  int idx = blockIdx.x * 256 + threadIdx.x;
  if (idx >= N * Kpad) return;
  int n = idx / Kpad, k = idx % Kpad;
  float scale = g[n] * rsqrtf(v[n] + 1e-5f);
  if (k == 0) bias[n] = b[n] - m[n] * scale;
  int K = C * KH * KW;
  float val = 0.f;
  if (k < K) {
    int s = k / C, ci = k % C;
    int ky = s / KW, kx = s % KW;
    val = w[((size_t)(n * C + ci) * KH + ky) * KW + kx] * scale;
  }
  wt[idx] = f2bf(val);
}

// ---------------------------------------------------------------------------
// Implicit-GEMM conv, BF16 WMMA, double-buffered LDS.
// Tile: 128(M pixels) x BNT(N ch) x 32(K). 8 waves in a 4x2 grid; each wave
// computes 32 x (BNT/2) -> 2 x (BNT/32) WMMA frags per k-step, serpentine
// issue order with A/B reuse hints. LDS rows use a 40-halfword stride (the
// TDM pad feature: 4 dwords of pad every 16 dwords) -> conflict-free 16B
// fragment loads. All accumulator loops are fully unrolled so the f32 frags
// stay in VGPRs (no scratch).
// ---------------------------------------------------------------------------
#define BM 128
#define BK 32
#define LDSK 40

struct ConvP {
  const unsigned short* act;   // NHWC bf16 input
  const unsigned short* wgt;   // [N][Kpad] bf16, BN-folded
  const float* bias;           // [N]
  const unsigned short* res;   // NHWC bf16 residual (or null)
  unsigned short* out_bf;      // NHWC bf16 out (or null)
  float* out_f32;              // NHWC f32 out (or null)
  int B, H, W, C, OH, OW, N, KH, KW, stride, pad, K, Kpad, relu;
};

__device__ __forceinline__ v16bf ld_frag(const unsigned short* base, int lane) {
  // ISA 16-bit A/B layout: lane&15 = row (M or N); lanes>=16 hold K+8.
  // vector elems 0..7 -> K kb..kb+7, elems 8..15 -> K kb+16..kb+23.
  const unsigned short* p = base + (lane & 15) * LDSK + ((lane >> 4) << 3);
  FragU u;
  u.q[0] = *reinterpret_cast<const B128*>(p);
  u.q[1] = *reinterpret_cast<const B128*>(p + 16);
  return u.v;
}

#define WMMA_BF16(A, Bv, C, RA, RB) \
  __builtin_amdgcn_wmma_f32_16x16x32_bf16(false, A, false, Bv, (short)0, C, RA, RB)

template <int BNT>
__global__ __launch_bounds__(256) void conv_wmma(ConvP p) {
  constexpr int WNT = BNT / 32;                 // n-tiles per wave
  __shared__ unsigned short As[2][BM * LDSK];   // 2 x 10 KB
  __shared__ unsigned short Bs[2][BNT * LDSK];  // 2 x (5|10) KB
  const int Mtot = p.B * p.OH * p.OW;
  const int m0 = blockIdx.x * BM;
  const int n0 = blockIdx.y * BNT;
  const int tid = threadIdx.x;
  const int lane = tid & 31;
  const int wave = tid >> 5;
  const int wm = (wave >> 1) << 5;              // 0,32,64,96
  const int wn = (wave & 1) * (BNT / 2);        // 0, BNT/2

  const v8f vzero = {0.f, 0.f, 0.f, 0.f, 0.f, 0.f, 0.f, 0.f};
  v8f acc[2][WNT];
#pragma unroll
  for (int i = 0; i < 2; ++i)
#pragma unroll
    for (int j = 0; j < WNT; ++j) acc[i][j] = vzero;

  const bool contig = (p.C % BK) == 0;

  // A-staging coords: 2 threads per row, 16 halfwords each
  const int arow = tid >> 1;
  const int ahalf = tid & 1;
  const int am = m0 + arow;
  const int ab = am / (p.OH * p.OW);
  const int arr = am % (p.OH * p.OW);
  const int aoy = arr / p.OW;
  const int aox = arr % p.OW;

  __builtin_prefetch(p.wgt + (size_t)n0 * p.Kpad, 0, 3);

  // ---- B tile: BNT rows (out ch) x 32 K, via Tensor Data Mover ----
  auto stageB = [&](int k0, int buf) {
#if defined(USE_TDM)
    if (tid == 0) {
      unsigned lds = (unsigned)(unsigned long long)(uintptr_t)(&Bs[buf][0]);
      unsigned long long ga =
          (unsigned long long)(uintptr_t)(p.wgt + (size_t)n0 * p.Kpad + k0);
      v4u g0;
      g0[0] = 1u;                                   // count=1 valid descriptor
      g0[1] = lds;                                  // lds_addr
      g0[2] = (unsigned)ga;                         // global_addr[31:0]
      g0[3] = ((unsigned)(ga >> 32) & 0x01FFFFFFu) | (2u << 30);  // addr[56:32], type=2
      v8i g1;
      // data_size=2B, pad_enable, pad_interval=16 dwords (code 3), pad_amount=4 dwords (code 3)
      g1[0] = (int)((1u << 16) | (1u << 20) | (3u << 22) | (3u << 25));
      g1[1] = (int)((unsigned)BK << 16);            // tensor_dim0 = 32  (bits 63:48)
      g1[2] = (int)((unsigned)BNT << 16);           // tensor_dim1 = BNT (bits 95:80)
      g1[3] = (int)((unsigned)BK << 16);            // tile_dim0 = 32    (bits 127:112)
      g1[4] = (int)BNT;                             // tile_dim1 = BNT
      g1[5] = p.Kpad;                               // tensor_dim0_stride
      g1[6] = 0; g1[7] = 0;
      v4i z4 = {0, 0, 0, 0};
#if __clang_major__ >= 23
      v8i z8 = {0, 0, 0, 0, 0, 0, 0, 0};
      __builtin_amdgcn_tensor_load_to_lds(g0, g1, z4, z4, z8, 0);
#else
      __builtin_amdgcn_tensor_load_to_lds(g0, g1, z4, z4, 0);
#endif
    }
#else
    for (int i = tid; i < BNT * 4; i += 256) {
      int r = i >> 2, q = i & 3;
      *reinterpret_cast<B128*>(&Bs[buf][r * LDSK + (q << 3)]) =
          *reinterpret_cast<const B128*>(p.wgt + (size_t)(n0 + r) * p.Kpad + k0 + (q << 3));
    }
#endif
  };

  // ---- A tile: im2col gather, zero-padded ----
  auto stageA = [&](int k0, int buf) {
    if (contig) {
      int s = k0 / p.C;
      int ci = (k0 % p.C) + (ahalf << 4);
      int ky = s / p.KW, kx = s % p.KW;
      int iy = aoy * p.stride - p.pad + ky;
      int ix = aox * p.stride - p.pad + kx;
      B128 v0 = {{0, 0, 0, 0}}, v1 = {{0, 0, 0, 0}};
      if (am < Mtot && iy >= 0 && iy < p.H && ix >= 0 && ix < p.W) {
        const unsigned short* sp =
            p.act + (((size_t)ab * p.H + iy) * p.W + ix) * p.C + ci;
        v0 = *reinterpret_cast<const B128*>(sp);
        v1 = *reinterpret_cast<const B128*>(sp + 8);
      }
      *reinterpret_cast<B128*>(&As[buf][arow * LDSK + (ahalf << 4)]) = v0;
      *reinterpret_cast<B128*>(&As[buf][arow * LDSK + (ahalf << 4) + 8]) = v1;
    } else {  // generic path (conv1: C=3, K=147)
      for (int j = 0; j < 16; ++j) {
        int kk = k0 + (ahalf << 4) + j;
        unsigned short v = 0;
        if (kk < p.K && am < Mtot) {
          int s = kk / p.C, ci = kk % p.C;
          int ky = s / p.KW, kx = s % p.KW;
          int iy = aoy * p.stride - p.pad + ky;
          int ix = aox * p.stride - p.pad + kx;
          if (iy >= 0 && iy < p.H && ix >= 0 && ix < p.W)
            v = p.act[(((size_t)ab * p.H + iy) * p.W + ix) * p.C + ci];
        }
        As[buf][arow * LDSK + (ahalf << 4) + j] = v;
      }
    }
  };

  const int nsteps = p.Kpad / BK;

  // prologue: fill buffer 0
  stageB(0, 0);
  stageA(0, 0);
#if defined(USE_TDM)
  __builtin_amdgcn_s_wait_tensorcnt(0);
#endif
  __syncthreads();

  for (int ks = 0; ks < nsteps; ++ks) {
    const int cur = ks & 1;
    // overlap: stage next tile into the other buffer while computing this one
    if (ks + 1 < nsteps) {
      stageB((ks + 1) * BK, cur ^ 1);
      stageA((ks + 1) * BK, cur ^ 1);
    }

    v16bf a0 = ld_frag(&As[cur][wm * LDSK], lane);
    v16bf a1 = ld_frag(&As[cur][(wm + 16) * LDSK], lane);
    v16bf bfr[WNT];
#pragma unroll
    for (int t = 0; t < WNT; ++t)
      bfr[t] = ld_frag(&Bs[cur][(wn + (t << 4)) * LDSK], lane);

    if constexpr (WNT == 2) {
      acc[0][0] = WMMA_BF16(a0, bfr[0], acc[0][0], true, false);
      acc[0][1] = WMMA_BF16(a0, bfr[1], acc[0][1], false, true);
      acc[1][1] = WMMA_BF16(a1, bfr[1], acc[1][1], true, false);
      acc[1][0] = WMMA_BF16(a1, bfr[0], acc[1][0], false, false);
    } else {
      acc[0][0] = WMMA_BF16(a0, bfr[0], acc[0][0], true, false);
      acc[0][1] = WMMA_BF16(a0, bfr[1], acc[0][1], true, false);
      acc[0][2] = WMMA_BF16(a0, bfr[2], acc[0][2], true, false);
      acc[0][3] = WMMA_BF16(a0, bfr[3], acc[0][3], false, true);
      acc[1][3] = WMMA_BF16(a1, bfr[3], acc[1][3], true, false);
      acc[1][2] = WMMA_BF16(a1, bfr[2], acc[1][2], true, false);
      acc[1][1] = WMMA_BF16(a1, bfr[1], acc[1][1], true, false);
      acc[1][0] = WMMA_BF16(a1, bfr[0], acc[1][0], false, false);
    }

#if defined(USE_TDM)
    __builtin_amdgcn_s_wait_tensorcnt(0);  // drain TDM issued this iteration
#endif
    __syncthreads();
  }

  // Epilogue: C/D layout -> lane&15 = N col, VGPR r = M row (+8 for lanes>=16)
  // Fully unrolled so acc indices are constants (keeps frags in VGPRs).
#pragma unroll
  for (int tm = 0; tm < 2; ++tm)
#pragma unroll
    for (int tn = 0; tn < WNT; ++tn) {
      int nn = n0 + wn + (tn << 4) + (lane & 15);
      int mb = m0 + wm + (tm << 4) + ((lane >> 4) << 3);
      float bs = p.bias[nn];
#pragma unroll
      for (int r = 0; r < 8; ++r) {
        int m = mb + r;
        if (m < Mtot) {
          float v = acc[tm][tn][r] + bs;
          size_t off = (size_t)m * p.N + nn;
          if (p.res) v += bf2f(p.res[off]);
          if (p.relu) v = fmaxf(v, 0.f);
          if (p.out_bf) p.out_bf[off] = f2bf(v);
          if (p.out_f32) p.out_f32[off] = v;
        }
      }
    }
}

// ---------------------------------------------------------------------------
// Bicubic warp of NCHW f32 input -> rotated NHWC bf16 (encoder input)
// ---------------------------------------------------------------------------
__global__ void warp_in(const float* __restrict__ img, unsigned short* __restrict__ out,
                        int B, int C, int H, int W, float cc, float ss) {
  int idx = blockIdx.x * 256 + threadIdx.x;
  if (idx >= B * H * W) return;
  int b = idx / (H * W), rp = idx % (H * W);
  int oy = rp / W, ox = rp % W;
  float gx = -1.f + 2.f * ox / (float)(W - 1);
  float gy = -1.f + 2.f * oy / (float)(H - 1);
  float g0 = cc * gx + ss * gy, g1 = -ss * gx + cc * gy;
  float x = (g0 + 1.f) * 0.5f * (W - 1), y = (g1 + 1.f) * 0.5f * (H - 1);
  float fx = floorf(x), fy = floorf(y);
  float wx[4], wy[4];
  cubic_w(x - fx, wx); cubic_w(y - fy, wy);
  int x0 = (int)fx, y0 = (int)fy;
  for (int c = 0; c < C; ++c) {
    float a = 0.f;
    for (int j = 0; j < 4; ++j) {
      int iy = y0 + j - 1;
      if (iy < 0 || iy >= H) continue;
      for (int i = 0; i < 4; ++i) {
        int ix = x0 + i - 1;
        if (ix < 0 || ix >= W) continue;
        a += wy[j] * wx[i] * img[((size_t)(b * C + c) * H + iy) * W + ix];
      }
    }
    out[(size_t)idx * C + c] = f2bf(a);
  }
}

// Bicubic back-warp of NHWC f32 features, max-accumulated over rotations
__global__ void warp_feat_max(const float* __restrict__ in, float* __restrict__ outmax,
                              int B, int H, int W, int C, float cc, float ss) {
  size_t idx = (size_t)blockIdx.x * 256 + threadIdx.x;
  size_t tot = (size_t)B * H * W * C;
  if (idx >= tot) return;
  int c = (int)(idx % C);
  size_t p1 = idx / C;
  int ox = (int)(p1 % W);
  size_t p2 = p1 / W;
  int oy = (int)(p2 % H);
  int b = (int)(p2 / H);
  float gx = -1.f + 2.f * ox / (float)(W - 1);
  float gy = -1.f + 2.f * oy / (float)(H - 1);
  float g0 = cc * gx + ss * gy, g1 = -ss * gx + cc * gy;
  float x = (g0 + 1.f) * 0.5f * (W - 1), y = (g1 + 1.f) * 0.5f * (H - 1);
  float fx = floorf(x), fy = floorf(y);
  float wx[4], wy[4];
  cubic_w(x - fx, wx); cubic_w(y - fy, wy);
  int x0 = (int)fx, y0 = (int)fy;
  float a = 0.f;
  for (int j = 0; j < 4; ++j) {
    int iy = y0 + j - 1;
    if (iy < 0 || iy >= H) continue;
    for (int i = 0; i < 4; ++i) {
      int ix = x0 + i - 1;
      if (ix < 0 || ix >= W) continue;
      a += wy[j] * wx[i] * in[((size_t)(b * H + iy) * W + ix) * C + c];
    }
  }
  outmax[idx] = fmaxf(outmax[idx], a);
}

// Identity-grid bicubic upsample: NHWC f32 -> NCHW f32
__global__ void upsample_nchw(const float* __restrict__ in, float* __restrict__ out,
                              int B, int Hi, int Wi, int C, int Ho, int Wo) {
  size_t idx = (size_t)blockIdx.x * 256 + threadIdx.x;
  size_t tot = (size_t)B * C * Ho * Wo;
  if (idx >= tot) return;
  int ox = (int)(idx % Wo);
  size_t p1 = idx / Wo;
  int oy = (int)(p1 % Ho);
  size_t p2 = p1 / Ho;
  int c = (int)(p2 % C);
  int b = (int)(p2 / C);
  float x = ox * (float)(Wi - 1) / (float)(Wo - 1);
  float y = oy * (float)(Hi - 1) / (float)(Ho - 1);
  float fx = floorf(x), fy = floorf(y);
  float wx[4], wy[4];
  cubic_w(x - fx, wx); cubic_w(y - fy, wy);
  int x0 = (int)fx, y0 = (int)fy;
  float a = 0.f;
  for (int j = 0; j < 4; ++j) {
    int iy = y0 + j - 1;
    if (iy < 0 || iy >= Hi) continue;
    for (int i = 0; i < 4; ++i) {
      int ix = x0 + i - 1;
      if (ix < 0 || ix >= Wi) continue;
      a += wy[j] * wx[i] * in[((size_t)(b * Hi + iy) * Wi + ix) * C + c];
    }
  }
  out[idx] = a;
}

__global__ void maxpool3s2(const unsigned short* __restrict__ in,
                           unsigned short* __restrict__ out, int B, int H, int W, int C) {
  int OH = (H + 2 - 3) / 2 + 1, OW = (W + 2 - 3) / 2 + 1;
  size_t idx = (size_t)blockIdx.x * 256 + threadIdx.x;
  size_t tot = (size_t)B * OH * OW * C;
  if (idx >= tot) return;
  int c = (int)(idx % C);
  size_t p1 = idx / C;
  int ox = (int)(p1 % OW);
  size_t p2 = p1 / OW;
  int oy = (int)(p2 % OH);
  int b = (int)(p2 / OH);
  float m = -__builtin_inff();
  for (int ky = 0; ky < 3; ++ky) {
    int iy = oy * 2 - 1 + ky;
    if (iy < 0 || iy >= H) continue;
    for (int kx = 0; kx < 3; ++kx) {
      int ix = ox * 2 - 1 + kx;
      if (ix < 0 || ix >= W) continue;
      m = fmaxf(m, bf2f(in[((size_t)(b * H + iy) * W + ix) * C + c]));
    }
  }
  out[idx] = f2bf(m);
}

__global__ void fill_f32(float* __restrict__ p, float v, size_t n) {
  size_t i = (size_t)blockIdx.x * 256 + threadIdx.x;
  if (i < n) p[i] = v;
}

// In-place channel l2norm on NCHW f32
__global__ void l2norm_nchw(float* __restrict__ x, int B, int C, int HW) {
  int idx = blockIdx.x * 256 + threadIdx.x;
  if (idx >= B * HW) return;
  int b = idx / HW, s = idx % HW;
  float* p = x + (size_t)b * C * HW + s;
  float ss = 0.f;
  for (int c = 0; c < C; ++c) { float v = p[(size_t)c * HW]; ss += v * v; }
  float inv = 1.f / fmaxf(sqrtf(ss), 1e-12f);
  for (int c = 0; c < C; ++c) p[(size_t)c * HW] *= inv;
}

// ---------------------------------------------------------------------------
// NetVLAD: soft-assign + accumulate (LDS-blocked), then normalize.
// x: NCHW f32 [N,128,S]; w: [64,128]; vlad: [N,64,128]; asum: [N,64]
// ---------------------------------------------------------------------------
__global__ __launch_bounds__(256) void vlad_acc(const float* __restrict__ x,
                                                const float* __restrict__ w,
                                                float* __restrict__ vlad,
                                                float* __restrict__ asum, int S) {
  __shared__ float lv[64 * 128];
  __shared__ float la[64];
  int n = blockIdx.y;
  for (int i = threadIdx.x; i < 64 * 128; i += 256) lv[i] = 0.f;
  if (threadIdx.x < 64) la[threadIdx.x] = 0.f;
  __syncthreads();
  int s = blockIdx.x * 256 + threadIdx.x;
  if (s < S) {
    const float* xp = x + (size_t)n * 128 * S + s;
    float lg[64];
    for (int k = 0; k < 64; ++k) lg[k] = 0.f;
    for (int c = 0; c < 128; ++c) {
      float xv = xp[(size_t)c * S];
      for (int k = 0; k < 64; ++k) lg[k] += w[k * 128 + c] * xv;
    }
    float mx = lg[0];
    for (int k = 1; k < 64; ++k) mx = fmaxf(mx, lg[k]);
    float sum = 0.f;
    for (int k = 0; k < 64; ++k) { lg[k] = expf(lg[k] - mx); sum += lg[k]; }
    float inv = 1.f / sum;
    for (int k = 0; k < 64; ++k) {
      float sa = lg[k] * inv;
      atomicAdd(&la[k], sa);
      for (int c = 0; c < 128; ++c)
        atomicAdd(&lv[k * 128 + c], sa * xp[(size_t)c * S]);
    }
  }
  __syncthreads();
  for (int i = threadIdx.x; i < 64 * 128; i += 256)
    atomicAdd(&vlad[(size_t)n * 8192 + i], lv[i]);
  if (threadIdx.x < 64) atomicAdd(&asum[n * 64 + threadIdx.x], la[threadIdx.x]);
}

__global__ __launch_bounds__(256) void vlad_fin(const float* __restrict__ vlad,
                                                const float* __restrict__ asum,
                                                const float* __restrict__ cent,
                                                float* __restrict__ out) {
  __shared__ float buf[8192];
  __shared__ float rn[64];
  __shared__ float gn;
  int n = blockIdx.x;
  if (threadIdx.x == 0) gn = 0.f;
  if (threadIdx.x < 64) rn[threadIdx.x] = 0.f;
  __syncthreads();
  for (int i = threadIdx.x; i < 8192; i += 256) {
    int k = i >> 7;
    float v = vlad[(size_t)n * 8192 + i] - asum[n * 64 + k] * cent[i];
    buf[i] = v;
    atomicAdd(&rn[k], v * v);
  }
  __syncthreads();
  for (int i = threadIdx.x; i < 8192; i += 256) {
    int k = i >> 7;
    float v = buf[i] / fmaxf(sqrtf(rn[k]), 1e-12f);
    buf[i] = v;
    atomicAdd(&gn, v * v);
  }
  __syncthreads();
  float inv = 1.f / fmaxf(sqrtf(gn), 1e-12f);
  for (int i = threadIdx.x; i < 8192; i += 256)
    out[(size_t)n * 8192 + i] = buf[i] * inv;
}

// ---------------------------------------------------------------------------
// Host orchestration.
// d_in index map (jax pytree / sorted-dict flattening of setup_inputs()):
//   0 centroids, 1 nv_w,
//   2..5  bn1{b,g,m,v}, 6 conv1,
//   7+10*i .. : layer1[i] {b1.b,g,m,v, b2.b,g,m,v, c1, c2}
//   37..51: layer2[0] {b1(4), b2(4), c1, c2, dbn(4), down}
//   52+10*(j-1): layer2[j] (j=1..3) {b1(4), b2(4), c1, c2}
//   81: x [4,3,224,224] f32
// ---------------------------------------------------------------------------
struct PW { unsigned short* wt; float* bias; int Kpad; };

extern "C" void kernel_launch(void* const* d_in, const int* in_sizes, int n_in,
                              void* d_out, int out_size, void* d_ws, size_t ws_size,
                              hipStream_t stream) {
  (void)in_sizes; (void)n_in; (void)out_size; (void)ws_size;
  const int B = 4;
  size_t off = 0;
  auto alloc = [&](size_t bytes) -> void* {
    void* p = (char*)d_ws + off;
    off += (bytes + 255) & ~(size_t)255;
    return p;
  };

  // ---- fold BN into BF16 weights (re-done every call: deterministic) ----
  auto prep = [&](int wi, int bb, int bg, int bm, int bv, int N, int C, int KH, int KW) -> PW {
    int K = C * KH * KW, Kpad = (K + 31) & ~31;
    PW r;
    r.wt = (unsigned short*)alloc((size_t)N * Kpad * 2);
    r.bias = (float*)alloc((size_t)N * 4);
    r.Kpad = Kpad;
    int tot = N * Kpad;
    prep_wgt<<<(tot + 255) / 256, 256, 0, stream>>>(
        (const float*)d_in[wi], (const float*)d_in[bg], (const float*)d_in[bb],
        (const float*)d_in[bm], (const float*)d_in[bv], r.wt, r.bias, N, C, KH, KW, Kpad);
    return r;
  };

  PW Wc1 = prep(6, 2, 3, 4, 5, 64, 3, 7, 7);
  PW Wl1[3][2];
  for (int i = 0; i < 3; ++i) {
    int bse = 7 + 10 * i;
    Wl1[i][0] = prep(bse + 8, bse + 0, bse + 1, bse + 2, bse + 3, 64, 64, 3, 3);
    Wl1[i][1] = prep(bse + 9, bse + 4, bse + 5, bse + 6, bse + 7, 64, 64, 3, 3);
  }
  PW Wl2[4][2], Wdn;
  Wl2[0][0] = prep(45, 37, 38, 39, 40, 128, 64, 3, 3);
  Wl2[0][1] = prep(46, 41, 42, 43, 44, 128, 128, 3, 3);
  Wdn       = prep(51, 47, 48, 49, 50, 128, 64, 1, 1);
  for (int j = 1; j < 4; ++j) {
    int bse = 52 + 10 * (j - 1);
    Wl2[j][0] = prep(bse + 8, bse + 0, bse + 1, bse + 2, bse + 3, 128, 128, 3, 3);
    Wl2[j][1] = prep(bse + 9, bse + 4, bse + 5, bse + 6, bse + 7, 128, 128, 3, 3);
  }

  // ---- activation buffers (NHWC) ----
  unsigned short* actIn = (unsigned short*)alloc((size_t)B * 224 * 224 * 3 * 2);
  unsigned short* c1o   = (unsigned short*)alloc((size_t)B * 112 * 112 * 64 * 2);
  unsigned short* pA = (unsigned short*)alloc((size_t)B * 56 * 56 * 64 * 2);
  unsigned short* pB = (unsigned short*)alloc((size_t)B * 56 * 56 * 64 * 2);
  unsigned short* pC = (unsigned short*)alloc((size_t)B * 56 * 56 * 64 * 2);
  unsigned short* qA = (unsigned short*)alloc((size_t)B * 28 * 28 * 128 * 2);
  unsigned short* qB = (unsigned short*)alloc((size_t)B * 28 * 28 * 128 * 2);
  unsigned short* qC = (unsigned short*)alloc((size_t)B * 28 * 28 * 128 * 2);
  unsigned short* qD = (unsigned short*)alloc((size_t)B * 28 * 28 * 128 * 2);
  float* feat = (float*)alloc((size_t)B * 28 * 28 * 128 * 4);
  float* equ  = (float*)alloc((size_t)B * 28 * 28 * 128 * 4);
  float* vlad = (float*)alloc((size_t)B * 64 * 128 * 4);
  float* asum = (float*)alloc((size_t)B * 64 * 4);

  auto conv = [&](const unsigned short* act, int H, int W, int C, const PW& pw,
                  int N, int KH, int KW, int stride, int pad,
                  const unsigned short* res, int relu,
                  unsigned short* obf, float* of32) {
    int OH = (H + 2 * pad - KH) / stride + 1;
    int OW = (W + 2 * pad - KW) / stride + 1;
    ConvP cp;
    cp.act = act; cp.wgt = pw.wt; cp.bias = pw.bias; cp.res = res;
    cp.out_bf = obf; cp.out_f32 = of32;
    cp.B = B; cp.H = H; cp.W = W; cp.C = C; cp.OH = OH; cp.OW = OW; cp.N = N;
    cp.KH = KH; cp.KW = KW; cp.stride = stride; cp.pad = pad;
    cp.K = C * KH * KW; cp.Kpad = pw.Kpad; cp.relu = relu;
    unsigned gx = (unsigned)((B * OH * OW + BM - 1) / BM);
    if (N % 128 == 0) {
      dim3 g(gx, (unsigned)(N / 128));
      conv_wmma<128><<<g, 256, 0, stream>>>(cp);
    } else {
      dim3 g(gx, (unsigned)(N / 64));
      conv_wmma<64><<<g, 256, 0, stream>>>(cp);
    }
  };

  const float* x_in = (const float*)d_in[81];
  const float PI = 3.14159265358979323846f;

  size_t equ_n = (size_t)B * 28 * 28 * 128;
  fill_f32<<<(unsigned)((equ_n + 255) / 256), 256, 0, stream>>>(equ, -__builtin_inff(), equ_n);

  for (int r = 0; r < 8; ++r) {
    float ang = (float)r * (PI / 4.f);        // = -angles[r]
    float c1r = cosf(ang), s1r = sinf(ang);   // forward warp: cos(-a), sin(-a)

    warp_in<<<(B * 224 * 224 + 255) / 256, 256, 0, stream>>>(
        x_in, actIn, B, 3, 224, 224, c1r, s1r);

    // encoder
    conv(actIn, 224, 224, 3, Wc1, 64, 7, 7, 2, 3, nullptr, 1, c1o, nullptr);
    {
      size_t tot = (size_t)B * 56 * 56 * 64;
      maxpool3s2<<<(unsigned)((tot + 255) / 256), 256, 0, stream>>>(c1o, pA, B, 112, 112, 64);
    }
    unsigned short *xp = pA, *tp = pB, *yp = pC;
    for (int i = 0; i < 3; ++i) {          // layer1
      conv(xp, 56, 56, 64, Wl1[i][0], 64, 3, 3, 1, 1, nullptr, 1, tp, nullptr);
      conv(tp, 56, 56, 64, Wl1[i][1], 64, 3, 3, 1, 1, xp, 1, yp, nullptr);
      unsigned short* sw = xp; xp = yp; yp = sw;
    }
    // layer2 block 0 (stride 2, projection shortcut)
    conv(xp, 56, 56, 64, Wdn, 128, 1, 1, 2, 0, nullptr, 0, qD, nullptr);
    conv(xp, 56, 56, 64, Wl2[0][0], 128, 3, 3, 2, 1, nullptr, 1, qA, nullptr);
    conv(qA, 28, 28, 128, Wl2[0][1], 128, 3, 3, 1, 1, qD, 1, qB, nullptr);
    unsigned short *x2 = qB, *t2 = qA, *y2 = qC;
    for (int j = 1; j < 4; ++j) {          // layer2 blocks 1..3
      conv(x2, 28, 28, 128, Wl2[j][0], 128, 3, 3, 1, 1, nullptr, 1, t2, nullptr);
      bool last = (j == 3);
      conv(t2, 28, 28, 128, Wl2[j][1], 128, 3, 3, 1, 1, x2, 1,
           last ? nullptr : y2, last ? feat : nullptr);
      if (!last) { unsigned short* sw = x2; x2 = y2; y2 = sw; }
    }

    // back-warp (cos(a), sin(a)) and max over rotations
    warp_feat_max<<<(unsigned)((equ_n + 255) / 256), 256, 0, stream>>>(
        feat, equ, B, 28, 28, 128, c1r, -s1r);
  }

  // ---- outputs ----
  float* out1  = (float*)d_out;                               // [4,128,56,56]
  float* local = out1 + (size_t)B * 128 * 56 * 56;            // [4,128,224,224]
  float* desc  = local + (size_t)B * 128 * 224 * 224;         // [4,8192]

  size_t n1 = (size_t)B * 128 * 56 * 56;
  upsample_nchw<<<(unsigned)((n1 + 255) / 256), 256, 0, stream>>>(equ, out1, B, 28, 28, 128, 56, 56);
  l2norm_nchw<<<(B * 56 * 56 + 255) / 256, 256, 0, stream>>>(out1, B, 128, 56 * 56);

  size_t n2 = (size_t)B * 128 * 224 * 224;
  upsample_nchw<<<(unsigned)((n2 + 255) / 256), 256, 0, stream>>>(equ, local, B, 28, 28, 128, 224, 224);
  l2norm_nchw<<<(B * 224 * 224 + 255) / 256, 256, 0, stream>>>(local, B, 128, 224 * 224);

  fill_f32<<<((B * 8192 + 255) / 256), 256, 0, stream>>>(vlad, 0.f, (size_t)B * 8192);
  fill_f32<<<1, 256, 0, stream>>>(asum, 0.f, (size_t)B * 64);
  {
    dim3 g((3136 + 255) / 256, B);
    vlad_acc<<<g, 256, 0, stream>>>(out1, (const float*)d_in[1], vlad, asum, 3136);
  }
  vlad_fin<<<B, 256, 0, stream>>>(vlad, asum, (const float*)d_in[0], desc);
}
